// LSTM_73254962200833
// MI455X (gfx1250) — matrix operand order repaired and checked by reading, not compiled, not run on hardware
//
#include <hip/hip_runtime.h>
#include <hip/hip_bf16.h>

// ---------------------------------------------------------------------------
// LSTM on MI455X (gfx1250), wave32, bf16 WMMA with f32 accumulation.
//   gates = [x_t | h_t] (B x 640) * W_cat^T (640 x 2048)  -- fused x-proj
//   persistent kernel, 512 steps, device-wide barrier between steps.
//   Entire per-wave weight slice (80 WMMA B-fragments = 640 VGPRs) is
//   register-resident for all 512 steps (CDNA5: 1024 VGPRs/wave); the
//   steady-state loop is pure WMMA + clause-batched, pipelined A loads.
//   Barrier sits between the (h-independent) x-projection and the h-GEMM.
//   Reuse-A hints removed: the scheduler reorders gate WMMAs, which breaks
//   the ISA legality condition for the hint (undefined results).
// ---------------------------------------------------------------------------

typedef __attribute__((ext_vector_type(16))) __bf16 v16bf;
typedef __attribute__((ext_vector_type(8)))  __bf16 v8bf;
typedef __attribute__((ext_vector_type(8)))  float  v8f;

#define T_STEPS 512
#define BATCH   256
#define NINP    128          // 127 input features + dt
#define NHID    512
#define NGATE   2048         // 4*NHID
#define KTOT    640          // NINP + NHID
#define NOUT    64
#define NBLK_LSTM 128        // (BATCH/64) * (NHID/16)

// round-to-nearest-even f32 -> bf16 (bit trick; no backend fptrunc dependency)
static __device__ __forceinline__ unsigned short f2bf(float f) {
    unsigned int u = __float_as_uint(f);
    u += 0x7FFFu + ((u >> 16) & 1u);
    return (unsigned short)(u >> 16);
}

static __device__ __forceinline__ v8bf ld8(const unsigned short* p) {
    return *(const v8bf*)p;   // 16B-aligned b128 load
}

static __device__ __forceinline__ v16bf join8(v8bf lo, v8bf hi) {
    union { v16bf v; v8bf h[2]; } u;
    u.h[0] = lo; u.h[1] = hi;
    return u.v;
}

static __device__ __forceinline__ float sigm(float x) {
    return 1.0f / (1.0f + __expf(-x));
}
static __device__ __forceinline__ float tanh_fast(float x) {
    float e = __expf(-2.0f * x);
    return (1.0f - e) / (1.0f + e);
}

// --------------------------- setup kernels ---------------------------------

__global__ void k_init(unsigned short* hbuf, unsigned* sync_ws) {
    int idx = blockIdx.x * blockDim.x + threadIdx.x;
    if (idx < 2 * BATCH * NHID) hbuf[idx] = 0;   // h0 = 0 (both buffers)
    if (idx < 64) sync_ws[idx] = 0;              // barrier state
}

// pack x = concat(input, dt) -> bf16 [T*B, 128]
__global__ void k_pack(const float* __restrict__ input, const float* __restrict__ dt,
                       unsigned short* __restrict__ xpack) {
    int idx = blockIdx.x * blockDim.x + threadIdx.x;   // < T*B*128
    int row = idx >> 7;
    int k   = idx & 127;
    float v = (k < NINP - 1) ? input[(size_t)row * (NINP - 1) + k] : dt[row];
    xpack[idx] = f2bf(v);
}

// W_cat bf16 [2048][640] = [W_ih | W_hh]; W_cls bf16 [64][512]; bsum = b_ih+b_hh
__global__ void k_weights(const float* __restrict__ W_ih, const float* __restrict__ W_hh,
                          const float* __restrict__ b_ih, const float* __restrict__ b_hh,
                          const float* __restrict__ W_cls,
                          unsigned short* __restrict__ wcat,
                          unsigned short* __restrict__ wcls,
                          float* __restrict__ bsum) {
    int idx = blockIdx.x * blockDim.x + threadIdx.x;
    const int R1 = NGATE * KTOT;
    const int R2 = R1 + NOUT * NHID;
    const int R3 = R2 + NGATE;
    if (idx < R1) {
        int g = idx / KTOT, k = idx % KTOT;
        float v = (k < NINP) ? W_ih[(size_t)g * NINP + k]
                             : W_hh[(size_t)g * NHID + (k - NINP)];
        wcat[idx] = f2bf(v);
    } else if (idx < R2) {
        int j = idx - R1;
        wcls[j] = f2bf(W_cls[j]);
    } else if (idx < R3) {
        int g = idx - R2;
        bsum[g] = b_ih[g] + b_hh[g];
    }
}

// --------------------------- recurrent kernel ------------------------------
// 128 blocks x 128 threads (4 waves, one per SIMD32). Wave owns a 16(batch)
// x 16(hidden) patch for all 4 gates and all 512 timesteps. Cell state and
// the complete weight slice live in VGPRs for the whole kernel.

// 4 gate WMMAs against register-resident B fragments for one k-step.
#define STEP_K(KS, AFRAG)                                                      \
    {                                                                          \
        v16bf a_ = (AFRAG);                                                    \
        acc[0] = __builtin_amdgcn_wmma_f32_16x16x32_bf16(                      \
            false, a_, false, bw[KS][0], (short)0, acc[0], false, false);      \
        acc[1] = __builtin_amdgcn_wmma_f32_16x16x32_bf16(                      \
            false, a_, false, bw[KS][1], (short)0, acc[1], false, false);      \
        acc[2] = __builtin_amdgcn_wmma_f32_16x16x32_bf16(                      \
            false, a_, false, bw[KS][2], (short)0, acc[2], false, false);      \
        acc[3] = __builtin_amdgcn_wmma_f32_16x16x32_bf16(                      \
            false, a_, false, bw[KS][3], (short)0, acc[3], false, false);      \
    }

__global__ void __launch_bounds__(128, 1)
k_lstm(const unsigned short* __restrict__ xpack,   // [T*B][128] bf16
       const unsigned short* __restrict__ wcat,    // [2048][640] bf16
       const float* __restrict__ bsum,             // [2048]
       unsigned short* __restrict__ hbuf,          // [2][B][NHID] bf16 double buffer
       unsigned short* __restrict__ hs,            // [T][B][NHID] bf16
       unsigned* __restrict__ sync_ws) {
    const int wave = threadIdx.x >> 5;
    const int lane = threadIdx.x & 31;
    const int lm   = lane & 15;       // A: row M, B: col N, C/D: col N
    const int hi   = lane >> 4;       // lane half selects K/M sub-range

    const int nb = blockIdx.x & 31;   // 32 hidden-column tiles
    const int mb = blockIdx.x >> 5;   // 4 batch groups of 64
    const int n0 = nb * 16;
    const int m0 = mb * 64 + wave * 16;

    // ---- one-time: load the wave's full weight slice into VGPRs ----
    // 20 k-steps x 4 gates x 8 VGPRs = 640 VGPRs (of 1024/wave on CDNA5).
    const unsigned short* wrow[4];
#pragma unroll
    for (int g = 0; g < 4; ++g)
        wrow[g] = wcat + (size_t)(g * NHID + n0 + lm) * KTOT + hi * 8;

    v16bf bw[20][4];
#pragma unroll
    for (int ks = 0; ks < 20; ++ks)
#pragma unroll
        for (int g = 0; g < 4; ++g)
            bw[ks][g] = join8(ld8(wrow[g] + ks * 32), ld8(wrow[g] + ks * 32 + 16));

    // per-gate bias (constant across steps), broadcast into accumulator init
    float bias[4];
#pragma unroll
    for (int g = 0; g < 4; ++g) bias[g] = bsum[g * NHID + n0 + lm];

    const int arow = m0 + lm;         // A-fragment batch row for this lane

    v8f creg = {};                    // cell state tile (f32, registers)

    for (int t = 0; t < T_STEPS; ++t) {
        const unsigned short* hcur = hbuf + (size_t)(t & 1) * BATCH * NHID;
        unsigned short*       hnxt = hbuf + (size_t)((t + 1) & 1) * BATCH * NHID;

        v8f acc[4];
#pragma unroll
        for (int g = 0; g < 4; ++g)
#pragma unroll
            for (int r = 0; r < 8; ++r) acc[g][r] = bias[g];

        // ---- x contribution: K = 0..127 -- independent of h, so it runs
        //      BEFORE the barrier and overlaps other blocks' step tail ----
        const unsigned short* ax = xpack + ((size_t)t * BATCH + arow) * NINP + hi * 8;
#pragma unroll
        for (int ks = 0; ks < 4; ++ks)
            STEP_K(ks, join8(ld8(ax + ks * 32), ld8(ax + ks * 32 + 16)))

        // ---- prefetch next step's activation rows (overlaps barrier wait) ----
        if (t + 1 < T_STEPS) {
            const char* pn =
                (const char*)(xpack + ((size_t)(t + 1) * BATCH + arow) * NINP);
            __builtin_prefetch(pn, 0, 0);         // global_prefetch_b8
            __builtin_prefetch(pn + 128, 0, 0);   // row = 256 B = 2 lines
        }

        // ---- device-wide barrier: publishes h written at the end of the
        //      previous iteration; x-part above is h-independent ----
        __syncthreads();
        if (threadIdx.x == 0) {
            __builtin_amdgcn_fence(__ATOMIC_RELEASE, "agent");
            unsigned prev = __hip_atomic_fetch_add(&sync_ws[0], 1u,
                                __ATOMIC_RELAXED, __HIP_MEMORY_SCOPE_AGENT);
            if (prev == (unsigned)(NBLK_LSTM * (t + 1) - 1)) {
                __hip_atomic_store(&sync_ws[1], (unsigned)(t + 1),
                                   __ATOMIC_RELAXED, __HIP_MEMORY_SCOPE_AGENT);
            } else {
                while (__hip_atomic_load(&sync_ws[1], __ATOMIC_RELAXED,
                                         __HIP_MEMORY_SCOPE_AGENT) < (unsigned)(t + 1)) {
                    __builtin_amdgcn_s_sleep(2);
                }
            }
        }
        __syncthreads();
        __builtin_amdgcn_fence(__ATOMIC_ACQUIRE, "agent");  // invalidate for fresh h

        // ---- h contribution: K = 128..639 (the true serialized path) ----
        const unsigned short* ah = hcur + (size_t)arow * NHID + hi * 8;
#pragma unroll
        for (int ks = 0; ks < 16; ++ks)
            STEP_K(4 + ks, join8(ld8(ah + ks * 32), ld8(ah + ks * 32 + 16)))

        // ---- gate nonlinearity + state update (wave-local) ----
        const int mrow = m0 + hi * 8;   // D layout: VGPR r -> batch mrow+r, col n0+lm
#pragma unroll
        for (int r = 0; r < 8; ++r) {
            float ig = sigm(acc[0][r]);
            float fg = sigm(acc[1][r]);
            float gg = tanh_fast(acc[2][r]);
            float og = sigm(acc[3][r]);
            float c  = fg * creg[r] + ig * gg;
            creg[r]  = c;
            float h  = og * tanh_fast(c);
            unsigned short hb = f2bf(h);
            hnxt[(size_t)(mrow + r) * NHID + n0 + lm] = hb;
            hs[((size_t)t * BATCH + mrow + r) * NHID + n0 + lm] = hb;
        }
    }
}

// --------------------------- classifier kernel -----------------------------
// out[T*B, 64] = hs(bf16) * W_cls^T(bf16) + b_cls, f32 out.

__global__ void __launch_bounds__(128)
k_cls(const unsigned short* __restrict__ hs,
      const unsigned short* __restrict__ wcls,
      const float* __restrict__ b_cls,
      float* __restrict__ out) {
    const int wave = threadIdx.x >> 5;
    const int lane = threadIdx.x & 31;
    const int lm = lane & 15, hi = lane >> 4;
    const int m0 = blockIdx.x * 64 + wave * 16;
    const int n0 = blockIdx.y * 16;

    float bias = b_cls[n0 + lm];
    v8f acc;
#pragma unroll
    for (int r = 0; r < 8; ++r) acc[r] = bias;

    const unsigned short* arow = hs   + (size_t)(m0 + lm) * NHID + hi * 8;
    const unsigned short* brow = wcls + (size_t)(n0 + lm) * NHID + hi * 8;
#pragma unroll 4
    for (int ks = 0; ks < 16; ++ks) {
        v16bf a = join8(ld8(arow + ks * 32), ld8(arow + ks * 32 + 16));
        v16bf b = join8(ld8(brow + ks * 32), ld8(brow + ks * 32 + 16));
        acc = __builtin_amdgcn_wmma_f32_16x16x32_bf16(
            false, a, false, b, (short)0, acc, false, false);
    }
    const int mrow = m0 + hi * 8;
#pragma unroll
    for (int r = 0; r < 8; ++r)
        out[(size_t)(mrow + r) * NOUT + n0 + lm] = acc[r];
}

// --------------------------- launch ----------------------------------------

extern "C" void kernel_launch(void* const* d_in, const int* in_sizes, int n_in,
                              void* d_out, int out_size, void* d_ws, size_t ws_size,
                              hipStream_t stream) {
    (void)in_sizes; (void)n_in; (void)out_size; (void)ws_size;

    const float* input = (const float*)d_in[0];
    const float* dt    = (const float*)d_in[1];
    const float* W_ih  = (const float*)d_in[2];
    const float* W_hh  = (const float*)d_in[3];
    const float* b_ih  = (const float*)d_in[4];
    const float* b_hh  = (const float*)d_in[5];
    const float* W_cls = (const float*)d_in[6];
    const float* b_cls = (const float*)d_in[7];

    char* ws = (char*)d_ws;                       // all offsets 256B aligned
    unsigned short* wcat  = (unsigned short*)(ws + 0);          //  2,621,440 B
    unsigned short* wcls  = (unsigned short*)(ws + 2621440);    //     65,536 B
    float*          bsum  = (float*)         (ws + 2686976);    //      8,192 B
    unsigned*       sync_ws = (unsigned*)    (ws + 2695168);    //        256 B
    unsigned short* hbuf  = (unsigned short*)(ws + 2695424);    //    524,288 B
    unsigned short* xpack = (unsigned short*)(ws + 3219712);    // 33,554,432 B
    unsigned short* hs    = (unsigned short*)(ws + 36774144);   // 134,217,728 B
                                                 // total ~163 MB

    k_init<<<1024, 256, 0, stream>>>(hbuf, sync_ws);
    k_pack<<<(T_STEPS * BATCH * NINP) / 256, 256, 0, stream>>>(input, dt, xpack);
    const int wtot = NGATE * KTOT + NOUT * NHID + NGATE;
    k_weights<<<(wtot + 255) / 256, 256, 0, stream>>>(W_ih, W_hh, b_ih, b_hh,
                                                      W_cls, wcat, wcls, bsum);
    k_lstm<<<NBLK_LSTM, 128, 0, stream>>>(xpack, wcat, bsum, hbuf, hs, sync_ws);
    k_cls<<<dim3((T_STEPS * BATCH) / 64, NOUT / 16), 128, 0, stream>>>(
        hs, wcls, b_cls, (float*)d_out);
}
